// SetConv1dDecoder_4569845203569
// MI455X (gfx1250) — matrix operand order repaired and checked by reading, not compiled, not run on hardware
//
#include <hip/hip_runtime.h>

typedef _Float16 f16;
typedef __attribute__((ext_vector_type(16))) _Float16 v16h;
typedef __attribute__((ext_vector_type(8)))  _Float16 v8h;
typedef __attribute__((ext_vector_type(4)))  _Float16 v4h;
typedef __attribute__((ext_vector_type(8)))  float    v8f;
typedef __attribute__((ext_vector_type(4)))  float    v4f;

#define B_   4
#define T_   4096
#define S_   4096
#define V_   512
#define BM   64
#define BN   256
#define BK   32
#define LDA  40   // padded row stride (f16) for ldsA
#define LDB  40   // padded column stride (f16) for ldsB
#define NSTEP (S_ / BK)   // 128

__global__ __launch_bounds__(256)
void setconv1d_decoder_kernel(const float* __restrict__ q,
                              const float* __restrict__ k,
                              const float* __restrict__ val,
                              const float* __restrict__ log_scale,
                              float* __restrict__ out)
{
    // double-buffered tiles: 2 * (5 KB + 20 KB) = 50 KB
    __shared__ __align__(16) f16 ldsA[2][BM * LDA];
    __shared__ __align__(16) f16 ldsB[2][BN * LDB];

    const int tid  = threadIdx.x;
    const int lane = tid & 31;
    const int wave = tid >> 5;
    const int wm   = wave & 1;    // 32-row (t) band within block
    const int wn   = wave >> 1;   // 64-col (v) band within block

    const int b  = blockIdx.z;
    const int m0 = blockIdx.x * BM;   // t offset
    const int n0 = blockIdx.y * BN;   // v offset

    const float c = -0.5f * __expf(-2.0f * log_scale[0]);

    const float* qb = q   + (size_t)b * T_;
    const float* kb = k   + (size_t)b * S_;
    const float* vb = val + (size_t)b * V_ * S_;

    // A-producer: each thread -> one t row (0..63), 8 consecutive s
    const int arow = tid >> 2;
    const int acol = (tid & 3) * 8;
    const float qv = qb[m0 + arow];

    // B-producer: each thread -> 4 consecutive s for one v row, 8 v-rows
    const int brow = tid >> 3;        // 0..31 (v within 32-row group)
    const int bcol = (tid & 7) * 4;   // s offset within tile

    v8f acc[2][4];
#pragma unroll
    for (int i = 0; i < 2; i++)
#pragma unroll
        for (int j = 0; j < 4; j++) acc[i][j] = (v8f){};

    const int mloc  = lane & 15;
    const int khalf = lane >> 4;      // 0 or 1

    // -------- software-pipeline stage registers (next tile's globals) ------
    v4f kst0, kst1;       // key[s] for A production
    v4f bst[8];           // value rows for B production

    // prefetch step 0
    kst0 = *(const v4f*)&kb[acol];
    kst1 = *(const v4f*)&kb[acol + 4];
#pragma unroll
    for (int i = 0; i < 8; i++)
        bst[i] = *(const v4f*)&vb[(size_t)(n0 + i * 32 + brow) * S_ + bcol];

    for (int step = 0; step < NSTEP; ++step) {
        const int p = step & 1;

        // ---- produce A tile from staged key regs: exp(c*(k-q)^2) as f16
        {
            v8h a;
#pragma unroll
            for (int j = 0; j < 4; j++) {
                float d0 = kst0[j] - qv;
                float d1 = kst1[j] - qv;
                a[j]     = (f16)__expf(c * d0 * d0);
                a[j + 4] = (f16)__expf(c * d1 * d1);
            }
            *(v8h*)&ldsA[p][arow * LDA + acol] = a;
        }

        // ---- produce B tile from staged value regs (f32 -> f16)
#pragma unroll
        for (int i = 0; i < 8; i++) {
            int n = i * 32 + brow;
            v4h h = { (f16)bst[i][0], (f16)bst[i][1],
                      (f16)bst[i][2], (f16)bst[i][3] };
            *(v4h*)&ldsB[p][n * LDB + bcol] = h;
        }

        __syncthreads();   // one barrier per step (double buffering)

        // ---- issue next step's global loads; latency hides under WMMA
        if (step + 1 < NSTEP) {
            const int kk = (step + 1) * BK;
            kst0 = *(const v4f*)&kb[kk + acol];
            kst1 = *(const v4f*)&kb[kk + acol + 4];
#pragma unroll
            for (int i = 0; i < 8; i++)
                bst[i] = *(const v4f*)&vb[(size_t)(n0 + i * 32 + brow) * S_ +
                                          kk + bcol];
        }

        // ---- A fragments (lanes 0-15: K={0..7,16..23}; lanes 16-31: +8)
        v16h afr[2];
#pragma unroll
        for (int mt = 0; mt < 2; mt++) {
            const f16* pa = &ldsA[p][(wm * 32 + mt * 16 + mloc) * LDA + khalf * 8];
            v8h lo = *(const v8h*)pa;
            v8h hi = *(const v8h*)(pa + 16);
            v16h t;
#pragma unroll
            for (int e = 0; e < 8; e++) { t[e] = lo[e]; t[e + 8] = hi[e]; }
            afr[mt] = t;
        }

        // ---- B fragments (lane = column n; lanes 0-15: K={0..7,16..23}; lanes 16-31: +8)
        v16h bfr[4];
#pragma unroll
        for (int nt = 0; nt < 4; nt++) {
            const f16* pb = &ldsB[p][(wn * 64 + nt * 16 + mloc) * LDB + khalf * 8];
            v8h lo = *(const v8h*)pb;
            v8h hi = *(const v8h*)(pb + 16);
            v16h t;
#pragma unroll
            for (int e = 0; e < 8; e++) { t[e] = lo[e]; t[e + 8] = hi[e]; }
            bfr[nt] = t;
        }

        // ---- 8 WMMA per wave per K-step
#pragma unroll
        for (int mt = 0; mt < 2; mt++)
#pragma unroll
            for (int nt = 0; nt < 4; nt++)
                acc[mt][nt] = __builtin_amdgcn_wmma_f32_16x16x32_f16(
                    false, afr[mt], false, bfr[nt],
                    (short)0, acc[mt][nt], false, false);
    }

    // ---- store C (non-temporal: out is write-once; keep value hot in L2)
    // VGPR r -> m = r + 8*(lane>=16), n = lane&15
#pragma unroll
    for (int mt = 0; mt < 2; mt++) {
#pragma unroll
        for (int nt = 0; nt < 4; nt++) {
            int vcol = n0 + wn * 64 + nt * 16 + mloc;
#pragma unroll
            for (int r = 0; r < 8; r++) {
                int trow = m0 + wm * 32 + mt * 16 + r + khalf * 8;
                __builtin_nontemporal_store(
                    acc[mt][nt][r],
                    &out[((size_t)b * T_ + trow) * V_ + vcol]);
            }
        }
    }
}

extern "C" void kernel_launch(void* const* d_in, const int* in_sizes, int n_in,
                              void* d_out, int out_size, void* d_ws, size_t ws_size,
                              hipStream_t stream) {
    const float* q   = (const float*)d_in[0];
    const float* k   = (const float*)d_in[1];
    const float* v   = (const float*)d_in[2];
    const float* ls  = (const float*)d_in[3];
    float* out       = (float*)d_out;

    dim3 grid(T_ / BM, V_ / BN, B_);   // 64 x 2 x 4
    setconv1d_decoder_kernel<<<grid, 256, 0, stream>>>(q, k, v, ls, out);
}
